// EncoderBlockWithMemory_33303176413708
// MI455X (gfx1250) — compile-verified, matrix operand order checked by reading, not run on hardware
//
#include <hip/hip_runtime.h>
#include <hip/hip_bf16.h>

// ---------------------------------------------------------------------------
// CDNA5 (gfx1250) transformer encoder block with memory tokens.
// - All heavy GEMMs: v_wmma_f32_16x16x32_bf16 (wave32 WMMA), f32 accumulate.
// - GEMM tile staging: global_load_async_to_lds_b128 (ASYNCcnt) with
//   double-buffered LDS tiles -> prefetch tile k+1 while computing tile k.
// ---------------------------------------------------------------------------

typedef __attribute__((ext_vector_type(16))) __bf16 v16bf;
typedef __attribute__((ext_vector_type(8)))  __bf16 v8bf;
typedef __attribute__((ext_vector_type(8)))  float  v8f;

static constexpr int B_    = 4;
static constexpr int S_    = 2048;
static constexpr int D_    = 1024;
static constexpr int H_    = 16;
static constexpr int DK_   = 64;
static constexpr int DFF_  = 4096;
static constexpr int MEMT_ = 32;
static constexpr int T_    = S_ + MEMT_;   // 2080 = 65 * 32

__device__ __forceinline__ v8f zero_v8f() {
  v8f z;
#pragma unroll
  for (int i = 0; i < 8; ++i) z[i] = 0.0f;
  return z;
}

// Build a 16-element bf16 A/B fragment from two contiguous 16-byte segments.
// Layout per CDNA5 ISA 7.12.2: lane holds row (lane%16); elements 0..7 are
// K = (lane/16)*8 .. +7, elements 8..15 are K = (lane/16)*8+16 .. +23.
__device__ __forceinline__ v16bf ld_frag(const __bf16* p0, const __bf16* p1) {
  v8bf lo = *(const v8bf*)p0;
  v8bf hi = *(const v8bf*)p1;
  return __builtin_shufflevector(lo, hi, 0, 1, 2, 3, 4, 5, 6, 7,
                                         8, 9, 10, 11, 12, 13, 14, 15);
}

__device__ __forceinline__ v8f wmma_bf16(v16bf a, v16bf b, v8f c) {
  return __builtin_amdgcn_wmma_f32_16x16x32_bf16(
      false, a, false, b, (short)0, c, false, false);
}

// CDNA5 async global->LDS copy, 16 bytes per lane (GVS mode: SGPR base +
// per-lane u32 byte offset). Tracked by ASYNCcnt; loads complete in order.
__device__ __forceinline__ void async_load_b128(unsigned lds_off,
                                                unsigned long long base,
                                                unsigned voff) {
  asm volatile("global_load_async_to_lds_b128 %0, %1, %2"
               :: "v"(lds_off), "v"(voff), "s"(base)
               : "memory");
}
__device__ __forceinline__ void wait_asynccnt_4() {
  asm volatile("s_wait_asynccnt 0x4" ::: "memory");
}
__device__ __forceinline__ void wait_asynccnt_0() {
  asm volatile("s_wait_asynccnt 0x0" ::: "memory");
}

// ---------------------------------------------------------------------------
// Weight transpose + fp32 -> bf16 convert: Wt[n*K + k] = (bf16)W[k*N + n]
// ---------------------------------------------------------------------------
__global__ __launch_bounds__(256)
void transpose_convert_kernel(const float* __restrict__ W,
                              __bf16* __restrict__ Wt, int K, int N) {
  int idx = blockIdx.x * 256 + threadIdx.x;
  int k = idx / N;
  int n = idx - k * N;
  Wt[(size_t)n * K + k] = (__bf16)W[idx];
}

__global__ __launch_bounds__(256)
void convert_kernel(const float* __restrict__ X, __bf16* __restrict__ Y, int n) {
  int i = blockIdx.x * 256 + threadIdx.x;
  if (i < n) Y[i] = (__bf16)X[i];
}

// ---------------------------------------------------------------------------
// LayerNorm (matches reference: unbiased std, alpha*(x-mean)/(std+eps)+bias)
// One block per row of D_=1024. Output bf16 for the following GEMM.
// ---------------------------------------------------------------------------
__global__ __launch_bounds__(256)
void layernorm_kernel(const float* __restrict__ x,
                      const float* __restrict__ alpha,
                      const float* __restrict__ beta,
                      __bf16* __restrict__ y) {
  __shared__ float red[256];
  const int row = blockIdx.x, tid = threadIdx.x;
  const float* xr = x + (size_t)row * D_;
  float vals[4];
  float s1 = 0.f, s2 = 0.f;
#pragma unroll
  for (int i = 0; i < 4; ++i) {
    float t = xr[tid + i * 256];
    vals[i] = t;
    s1 += t;
    s2 += t * t;
  }
  red[tid] = s1; __syncthreads();
  for (int o = 128; o > 0; o >>= 1) {
    if (tid < o) red[tid] += red[tid + o];
    __syncthreads();
  }
  float sum1 = red[0]; __syncthreads();
  red[tid] = s2; __syncthreads();
  for (int o = 128; o > 0; o >>= 1) {
    if (tid < o) red[tid] += red[tid + o];
    __syncthreads();
  }
  float sum2 = red[0];
  float mean = sum1 * (1.0f / D_);
  float var = (sum2 - (float)D_ * mean * mean) * (1.0f / (D_ - 1));
  var = var > 0.f ? var : 0.f;
  float rinv = 1.0f / (sqrtf(var) + 1e-6f);
  __bf16* yr = y + (size_t)row * D_;
#pragma unroll
  for (int i = 0; i < 4; ++i) {
    int d = tid + i * 256;
    yr[d] = (__bf16)(alpha[d] * (vals[i] - mean) * rinv + beta[d]);
  }
}

// ---------------------------------------------------------------------------
// WMMA GEMM: C[M,N] = A[M,K](bf16) * Bt[N,K](bf16)^T  (+ epilogue)
// MODE 0: store bf16
// MODE 1: +bias[col], ReLU, store bf16
// MODE 2: +resid[row,col], store f32
// MODE 3: +bias[col] + resid[row,col], store f32
// Block tile 128x128, K-step 32. 8 waves: 4 (M) x 2 (N); wave tile 32x64.
// Tiles staged with async global->LDS b128 copies, double-buffered so the
// next K-tile streams in while WMMAs consume the current one.
// Rows >= M are clamped (not skipped) so every wave issues exactly 4 async
// ops per tile -> ASYNCcnt bookkeeping is uniform; epilogue guards stores.
// ---------------------------------------------------------------------------
template <int MODE>
__global__ __launch_bounds__(256)
void gemm_bt_kernel(const __bf16* __restrict__ A, const __bf16* __restrict__ Bt,
                    const float* __restrict__ bias, const float* __restrict__ resid,
                    float* __restrict__ Cf, __bf16* __restrict__ Cb,
                    int M, int N, int K) {
  __shared__ __align__(16) __bf16 sA[2][128 * 32];
  __shared__ __align__(16) __bf16 sB[2][128 * 32];
  const int tid = threadIdx.x;
  const int lane = tid & 31, wave = tid >> 5;
  const int mw = wave >> 1, nw = wave & 1;
  const int mbase = blockIdx.y * 128, nbase = blockIdx.x * 128;
  const int lrow = lane & 15, lh = lane >> 4;

  const unsigned long long baseA = (unsigned long long)(uintptr_t)A;
  const unsigned long long baseB = (unsigned long long)(uintptr_t)Bt;

  // this thread's two 16B-segment assignments within the 128x32 tile
  int r0 = tid >> 2, c0 = (tid & 3) * 8;
  int idx1 = tid + 256;
  int r1 = idx1 >> 2, c1 = (idx1 & 3) * 8;
  int ga0 = mbase + r0; if (ga0 >= M) ga0 = M - 1;   // clamp, never skip
  int ga1 = mbase + r1; if (ga1 >= M) ga1 = M - 1;

  auto issue_tile = [&](int kk, int p) {
    async_load_b128((unsigned)(uintptr_t)&sA[p][r0 * 32 + c0], baseA,
                    (unsigned)(((size_t)ga0 * K + kk + c0) * 2));
    async_load_b128((unsigned)(uintptr_t)&sB[p][r0 * 32 + c0], baseB,
                    (unsigned)(((size_t)(nbase + r0) * K + kk + c0) * 2));
    async_load_b128((unsigned)(uintptr_t)&sA[p][r1 * 32 + c1], baseA,
                    (unsigned)(((size_t)ga1 * K + kk + c1) * 2));
    async_load_b128((unsigned)(uintptr_t)&sB[p][r1 * 32 + c1], baseB,
                    (unsigned)(((size_t)(nbase + r1) * K + kk + c1) * 2));
  };

  v8f acc[2][4];
#pragma unroll
  for (int mi = 0; mi < 2; ++mi)
#pragma unroll
    for (int ni = 0; ni < 4; ++ni) acc[mi][ni] = zero_v8f();

  issue_tile(0, 0);
  int p = 0;
  for (int kk = 0; kk < K; kk += 32) {
    const bool pf = (kk + 32) < K;
    if (pf) {
      issue_tile(kk + 32, p ^ 1);   // 4 new in flight on top of 4 old
      wait_asynccnt_4();            // in-order completion: tile p has landed
    } else {
      wait_asynccnt_0();
    }
    __syncthreads();

    v16bf af[2], bfr[4];
#pragma unroll
    for (int mi = 0; mi < 2; ++mi) {
      const __bf16* q = sA[p] + (mw * 32 + mi * 16 + lrow) * 32 + lh * 8;
      af[mi] = ld_frag(q, q + 16);
    }
#pragma unroll
    for (int ni = 0; ni < 4; ++ni) {
      const __bf16* q = sB[p] + (nw * 64 + ni * 16 + lrow) * 32 + lh * 8;
      bfr[ni] = ld_frag(q, q + 16);
    }
#pragma unroll
    for (int mi = 0; mi < 2; ++mi)
#pragma unroll
      for (int ni = 0; ni < 4; ++ni)
        acc[mi][ni] = wmma_bf16(af[mi], bfr[ni], acc[mi][ni]);

    __syncthreads();                // tile p free for overwrite next round
    p ^= 1;
  }

  // epilogue (C layout: VGPR j -> row j + 8*(lane/16); col = lane%16)
#pragma unroll
  for (int mi = 0; mi < 2; ++mi) {
#pragma unroll
    for (int ni = 0; ni < 4; ++ni) {
      int col = nbase + nw * 64 + ni * 16 + lrow;
#pragma unroll
      for (int j = 0; j < 8; ++j) {
        int row = mbase + mw * 32 + mi * 16 + j + 8 * lh;
        if (row >= M) continue;
        float r = acc[mi][ni][j];
        if (MODE == 1 || MODE == 3) r += bias[col];
        if (MODE == 1) r = r > 0.f ? r : 0.f;
        if (MODE == 2 || MODE == 3) r += resid[(size_t)row * N + col];
        if (MODE == 0 || MODE == 1)
          Cb[(size_t)row * N + col] = (__bf16)r;
        else
          Cf[(size_t)row * N + col] = r;
      }
    }
  }
}

// ---------------------------------------------------------------------------
// Flash attention with memory K/V prefix (T = 32 + 2048).
// Grid: (S/128, B*H). 8 waves/block, each wave owns 16 query rows.
// Online softmax; ctx accumulated in 4x v8f per wave (16 rows x 64 dims).
// ---------------------------------------------------------------------------
__global__ __launch_bounds__(256)
void attention_kernel(const __bf16* __restrict__ qg, const __bf16* __restrict__ kg,
                      const __bf16* __restrict__ vg,
                      const __bf16* __restrict__ memk, const __bf16* __restrict__ memv,
                      const int* __restrict__ mask,
                      __bf16* __restrict__ ctxout) {
  __shared__ __align__(16) __bf16 sVt[64 * 32];        // V tile transposed [d][key]
  __shared__ __align__(16) __bf16 sP[8 * 16 * 32];     // per-wave P tiles

  const int tid = threadIdx.x, lane = tid & 31, wave = tid >> 5;
  const int lrow = lane & 15, lh = lane >> 4;
  const int bh = blockIdx.y;
  const int b = bh >> 4, h = bh & 15;
  const int qbase = blockIdx.x * 128 + wave * 16;

  // Q fragments (rows qbase..+15, d 0..63), pre-scaled by 1/sqrt(64)=0.125
  v16bf qf[2];
  {
    const __bf16* qp =
        qg + ((size_t)b * S_ + qbase + lrow) * D_ + h * DK_ + lh * 8;
#pragma unroll
    for (int half = 0; half < 2; ++half) {
      v16bf f = ld_frag(qp + half * 32, qp + half * 32 + 16);
#pragma unroll
      for (int i = 0; i < 16; ++i) f[i] = (__bf16)((float)f[i] * 0.125f);
      qf[half] = f;
    }
  }

  v8f ctx[4];
#pragma unroll
  for (int dg = 0; dg < 4; ++dg) ctx[dg] = zero_v8f();
  float mrun[8], lsum[8];
#pragma unroll
  for (int j = 0; j < 8; ++j) { mrun[j] = -3.0e38f; lsum[j] = 0.f; }

  for (int tbase = 0; tbase < T_; tbase += 32) {
    __syncthreads();
    // stage transposed V tile (32 keys x 64 dims) cooperatively
    {
      int key = tid >> 3;
      int dseg = (tid & 7) * 8;
      int tg = tbase + key;
      const __bf16* src =
          (tg < MEMT_) ? (memv + (size_t)tg * D_ + h * DK_ + dseg)
                       : (vg + ((size_t)b * S_ + (tg - MEMT_)) * D_ + h * DK_ + dseg);
      v8bf val = *(const v8bf*)src;
#pragma unroll
      for (int i = 0; i < 8; ++i) sVt[(dseg + i) * 32 + key] = val[i];
    }
    __syncthreads();

    // scores: [16 rows x 32 keys] = Q[16x64] @ K^T
    v8f s0 = zero_v8f(), s1 = zero_v8f();
#pragma unroll
    for (int grp = 0; grp < 2; ++grp) {
      int tg = tbase + grp * 16 + lrow;
      const __bf16* kp =
          (tg < MEMT_) ? (memk + (size_t)tg * D_ + h * DK_)
                       : (kg + ((size_t)b * S_ + (tg - MEMT_)) * D_ + h * DK_);
      v16bf kf0 = ld_frag(kp + lh * 8, kp + lh * 8 + 16);
      v16bf kf1 = ld_frag(kp + 32 + lh * 8, kp + 32 + lh * 8 + 16);
      if (grp == 0) {
        s0 = wmma_bf16(qf[0], kf0, s0);
        s0 = wmma_bf16(qf[1], kf1, s0);
      } else {
        s1 = wmma_bf16(qf[0], kf0, s1);
        s1 = wmma_bf16(qf[1], kf1, s1);
      }
    }

    // online softmax: per-lane rows j + 8*(lane/16); 16-lane reductions
#pragma unroll
    for (int j = 0; j < 8; ++j) {
      int srow = qbase + j + 8 * lh;
      float v0 = mask[(size_t)srow * T_ + tbase + lrow] ? s0[j] : -1e9f;
      float v1 = mask[(size_t)srow * T_ + tbase + 16 + lrow] ? s1[j] : -1e9f;
      float tmax = fmaxf(v0, v1);
#pragma unroll
      for (int o = 8; o >= 1; o >>= 1) tmax = fmaxf(tmax, __shfl_xor(tmax, o, 32));
      float newm = fmaxf(mrun[j], tmax);
      float scale = __expf(mrun[j] - newm);
      float p0 = __expf(v0 - newm);
      float p1 = __expf(v1 - newm);
      float ts = p0 + p1;
#pragma unroll
      for (int o = 8; o >= 1; o >>= 1) ts += __shfl_xor(ts, o, 32);
      lsum[j] = lsum[j] * scale + ts;
      mrun[j] = newm;
#pragma unroll
      for (int dg = 0; dg < 4; ++dg) ctx[dg][j] *= scale;
      __bf16* pp = sP + wave * 512 + (j + 8 * lh) * 32;
      pp[lrow] = (__bf16)p0;
      pp[16 + lrow] = (__bf16)p1;
    }

    // P (16x32) as A fragment from wave-private LDS (same-wave RAW, in-order DS)
    const __bf16* pb = sP + wave * 512 + lrow * 32 + lh * 8;
    v16bf pf = ld_frag(pb, pb + 16);

    // ctx += P @ V  (N = 64 dims, K = 32 keys)
#pragma unroll
    for (int dg = 0; dg < 4; ++dg) {
      const __bf16* vp = sVt + (dg * 16 + lrow) * 32 + lh * 8;
      v16bf vf = ld_frag(vp, vp + 16);
      ctx[dg] = wmma_bf16(pf, vf, ctx[dg]);
    }
  }

  // normalize and write ctx as bf16 in [B,S,D] (head-major cols)
  float inv[8];
#pragma unroll
  for (int j = 0; j < 8; ++j) inv[j] = 1.0f / lsum[j];
#pragma unroll
  for (int dg = 0; dg < 4; ++dg)
#pragma unroll
    for (int j = 0; j < 8; ++j) {
      int srow = qbase + j + 8 * lh;
      int col = h * DK_ + dg * 16 + lrow;
      ctxout[((size_t)b * S_ + srow) * D_ + col] = (__bf16)(ctx[dg][j] * inv[j]);
    }
}

// ---------------------------------------------------------------------------
// new_memory: broadcast mean over S of final x into [B, MEM, D]
// ---------------------------------------------------------------------------
__global__ __launch_bounds__(256)
void mean_bcast_kernel(const float* __restrict__ x, float* __restrict__ nm) {
  int gid = blockIdx.x * 256 + threadIdx.x;    // B*D threads
  int b = gid / D_;
  int d = gid - b * D_;
  const float* p = x + (size_t)b * S_ * D_ + d;
  float s = 0.f;
  for (int i = 0; i < S_; ++i) s += p[(size_t)i * D_];
  float mean = s * (1.0f / S_);
  float* o = nm + (size_t)b * MEMT_ * D_ + d;
  for (int mI = 0; mI < MEMT_; ++mI) o[(size_t)mI * D_] = mean;
}

// ---------------------------------------------------------------------------
// launch
// ---------------------------------------------------------------------------
extern "C" void kernel_launch(void* const* d_in, const int* in_sizes, int n_in,
                              void* d_out, int out_size, void* d_ws, size_t ws_size,
                              hipStream_t stream) {
  (void)in_sizes; (void)n_in; (void)out_size; (void)ws_size;
  const float* x    = (const float*)d_in[0];
  const int*   mask = (const int*)d_in[1];
  const float* mem  = (const float*)d_in[2];
  const float* w_q  = (const float*)d_in[3];
  const float* w_k  = (const float*)d_in[4];
  const float* w_v  = (const float*)d_in[5];
  const float* w_o  = (const float*)d_in[6];
  const float* ln1a = (const float*)d_in[7];
  const float* ln1b = (const float*)d_in[8];
  const float* ln2a = (const float*)d_in[9];
  const float* ln2b = (const float*)d_in[10];
  const float* fw1  = (const float*)d_in[11];
  const float* fb1  = (const float*)d_in[12];
  const float* fw2  = (const float*)d_in[13];
  const float* fb2  = (const float*)d_in[14];

  float* out_x  = (float*)d_out;                                // [B,S,D]
  float* out_nm = out_x + (size_t)B_ * S_ * D_;                 // [B,MEM,D]

  // workspace carve-out
  size_t off = 0;
  auto carve = [&](size_t bytes) -> char* {
    char* p = (char*)d_ws + off;
    off = (off + bytes + 255) & ~(size_t)255;
    return p;
  };
  __bf16* wqT  = (__bf16*)carve((size_t)D_ * D_ * 2);
  __bf16* wkT  = (__bf16*)carve((size_t)D_ * D_ * 2);
  __bf16* wvT  = (__bf16*)carve((size_t)D_ * D_ * 2);
  __bf16* woT  = (__bf16*)carve((size_t)D_ * D_ * 2);
  __bf16* w1T  = (__bf16*)carve((size_t)DFF_ * D_ * 2);
  __bf16* w2T  = (__bf16*)carve((size_t)D_ * DFF_ * 2);
  __bf16* memb = (__bf16*)carve((size_t)MEMT_ * D_ * 2);
  __bf16* memK = (__bf16*)carve((size_t)MEMT_ * D_ * 2);
  __bf16* memV = (__bf16*)carve((size_t)MEMT_ * D_ * 2);
  __bf16* xn   = (__bf16*)carve((size_t)B_ * S_ * D_ * 2);
  __bf16* qb   = (__bf16*)carve((size_t)B_ * S_ * D_ * 2);
  __bf16* kb   = (__bf16*)carve((size_t)B_ * S_ * D_ * 2);
  __bf16* vb   = (__bf16*)carve((size_t)B_ * S_ * D_ * 2);
  __bf16* ctxb = (__bf16*)carve((size_t)B_ * S_ * D_ * 2);
  float*  x1   = (float*)carve((size_t)B_ * S_ * D_ * 4);
  __bf16* xn2  = (__bf16*)carve((size_t)B_ * S_ * D_ * 2);
  __bf16* ffh  = (__bf16*)carve((size_t)B_ * S_ * DFF_ * 2);

  const int rows = B_ * S_;   // 8192
  dim3 blk(256);

  // weight prep (bf16, transposed to [N][K])
  transpose_convert_kernel<<<(D_ * D_) / 256, blk, 0, stream>>>(w_q, wqT, D_, D_);
  transpose_convert_kernel<<<(D_ * D_) / 256, blk, 0, stream>>>(w_k, wkT, D_, D_);
  transpose_convert_kernel<<<(D_ * D_) / 256, blk, 0, stream>>>(w_v, wvT, D_, D_);
  transpose_convert_kernel<<<(D_ * D_) / 256, blk, 0, stream>>>(w_o, woT, D_, D_);
  transpose_convert_kernel<<<(D_ * DFF_) / 256, blk, 0, stream>>>(fw1, w1T, D_, DFF_);
  transpose_convert_kernel<<<(DFF_ * D_) / 256, blk, 0, stream>>>(fw2, w2T, DFF_, D_);
  convert_kernel<<<(MEMT_ * D_) / 256, blk, 0, stream>>>(mem, memb, MEMT_ * D_);

  // LN1 -> Q/K/V projections (WMMA bf16)
  layernorm_kernel<<<rows, blk, 0, stream>>>(x, ln1a, ln1b, xn);
  gemm_bt_kernel<0><<<dim3(D_ / 128, rows / 128), blk, 0, stream>>>(
      xn, wqT, nullptr, nullptr, nullptr, qb, rows, D_, D_);
  gemm_bt_kernel<0><<<dim3(D_ / 128, rows / 128), blk, 0, stream>>>(
      xn, wkT, nullptr, nullptr, nullptr, kb, rows, D_, D_);
  gemm_bt_kernel<0><<<dim3(D_ / 128, rows / 128), blk, 0, stream>>>(
      xn, wvT, nullptr, nullptr, nullptr, vb, rows, D_, D_);
  // memory K/V (raw memory tokens, not layernormed), M=32 with row clamp
  gemm_bt_kernel<0><<<dim3(D_ / 128, 1), blk, 0, stream>>>(
      memb, wkT, nullptr, nullptr, nullptr, memK, MEMT_, D_, D_);
  gemm_bt_kernel<0><<<dim3(D_ / 128, 1), blk, 0, stream>>>(
      memb, wvT, nullptr, nullptr, nullptr, memV, MEMT_, D_, D_);

  // flash attention over T = MEM + S
  attention_kernel<<<dim3(S_ / 128, B_ * H_), blk, 0, stream>>>(
      qb, kb, vb, memK, memV, mask, ctxb);

  // out-proj + residual -> x1 (f32)
  gemm_bt_kernel<2><<<dim3(D_ / 128, rows / 128), blk, 0, stream>>>(
      ctxb, woT, nullptr, x, x1, nullptr, rows, D_, D_);

  // LN2 -> FFN
  layernorm_kernel<<<rows, blk, 0, stream>>>(x1, ln2a, ln2b, xn2);
  gemm_bt_kernel<1><<<dim3(DFF_ / 128, rows / 128), blk, 0, stream>>>(
      xn2, w1T, fb1, nullptr, nullptr, ffh, rows, DFF_, D_);
  gemm_bt_kernel<3><<<dim3(D_ / 128, rows / 128), blk, 0, stream>>>(
      ffh, w2T, fb2, x1, out_x, nullptr, rows, D_, DFF_);

  // new_memory = broadcast mean over S
  mean_bcast_kernel<<<(B_ * D_) / 256, blk, 0, stream>>>(out_x, out_nm);
}
